// GCNModelVAE_31224412242681
// MI455X (gfx1250) — matrix operand order repaired and checked by reading, not compiled, not run on hardware
//
#include <hip/hip_runtime.h>
#include <hip/hip_bf16.h>

// ---------------------------------------------------------------------------
// GCN-VAE forward for MI455X (gfx1250), bf16 WMMA with f32 accumulation.
//   supportT = (x @ Wh)^T            bf16 [256,8192]
//   hidden   = relu(adj @ support)   bf16 [8192,256]
//   zsupT    = (hidden @ [Wm|Wl])^T  bf16 [128,8192]
//   zcat     = adj @ zsup            f32  [8192,128]  (cols 0:64 = z_mean)
//   recon    = z_mean @ z_mean^T     f32  [8192,8192]
// d_out = recon | z_mean | z_log_std
// Transposed bf16 intermediates => next GEMM's B tile staging is a pure
// 32B/lane copy (async global->LDS, ASYNCcnt) with no transpose scatter.
// ---------------------------------------------------------------------------

typedef __bf16        bf16_t;
typedef float         v8f   __attribute__((ext_vector_type(8)));
typedef bf16_t        v16bf __attribute__((ext_vector_type(16)));
typedef float         f32x4 __attribute__((ext_vector_type(4)));
typedef unsigned int  u32x4 __attribute__((ext_vector_type(4)));

#if __has_builtin(__builtin_amdgcn_global_load_async_to_lds_b128) && \
    __has_builtin(__builtin_amdgcn_s_wait_asynccnt)
#define USE_ASYNC_LDS 1
// Builtin expects: (v4i __device__* src, v4i __shared__* dst, imm offset, imm cpol)
typedef int v4i_t __attribute__((vector_size(16)));
typedef __attribute__((address_space(1))) v4i_t* g_v4i_p;
typedef __attribute__((address_space(3))) v4i_t* l_v4i_p;
#endif

#define BM 128
#define BN 64
#define BK 32
#define LDS_STRIDE 40   // shorts: 32 data + 8 pad (80 B rows) -> conflict-free frag reads

// ---- f32 -> packed 2x bf16 ------------------------------------------------
static __device__ __forceinline__ unsigned pack2(float lo, float hi) {
#if __has_builtin(__builtin_amdgcn_cvt_pk_bf16_f32)
    return __builtin_bit_cast(unsigned, __builtin_amdgcn_cvt_pk_bf16_f32(lo, hi));
#else
    unsigned a = __builtin_bit_cast(unsigned, lo) + 0x8000u;   // round-half-up
    unsigned b = __builtin_bit_cast(unsigned, hi) + 0x8000u;
    return __builtin_amdgcn_perm(b, a, 0x07060302u);           // {b.hi16, a.hi16}
#endif
}

static __device__ __forceinline__ unsigned short f32_to_bf16(float f) {
    return (unsigned short)(pack2(f, 0.0f) & 0xffffu);
}

// Load 16 consecutive f32 from global, convert, store 32 B to LDS.
static __device__ __forceinline__ void cvt_store16(const f32x4* __restrict__ g,
                                                   unsigned short* __restrict__ l) {
    f32x4 f0 = g[0], f1 = g[1], f2 = g[2], f3 = g[3];
    u32x4 q0, q1;
    q0.x = pack2(f0.x, f0.y); q0.y = pack2(f0.z, f0.w);
    q0.z = pack2(f1.x, f1.y); q0.w = pack2(f1.z, f1.w);
    q1.x = pack2(f2.x, f2.y); q1.y = pack2(f2.z, f2.w);
    q1.z = pack2(f3.x, f3.y); q1.w = pack2(f3.z, f3.w);
    u32x4* d = reinterpret_cast<u32x4*>(l);
    d[0] = q0; d[1] = q1;
}

// Copy 32 B of bf16 global -> LDS (async path uses ASYNCcnt).
static __device__ __forceinline__ void stage32B(const unsigned short* g,
                                                unsigned short* l) {
#ifdef USE_ASYNC_LDS
    __builtin_amdgcn_global_load_async_to_lds_b128((g_v4i_p)g,       (l_v4i_p)l,       0, 0);
    __builtin_amdgcn_global_load_async_to_lds_b128((g_v4i_p)(g + 8), (l_v4i_p)(l + 8), 0, 0);
#else
    const u32x4* gs = reinterpret_cast<const u32x4*>(g);
    u32x4 a = gs[0], b = gs[1];
    u32x4* d = reinterpret_cast<u32x4*>(l);
    d[0] = a; d[1] = b;
#endif
}

union Frag { u32x4 q[2]; v16bf v; };

enum { A_F32 = 0, A_BF16 = 1 };
enum { B_F32 = 0, B_BF16T = 1 };                 // B_F32: row-major [KxN] f32
enum { OUT_F32 = 0, OUT_BF16_ROW_RELU = 1, OUT_BF16_TRANS = 2 };

// C = op(A @ B).  A: row-major [MxK] (lda).  B_BF16T: bf16 [NxK] (ldb).
// OUT_BF16_TRANS writes C^T bf16 [NxM] with leading dim ldc (=M).
template <int AM, int BMODE, int OM>
__global__ __launch_bounds__(256) void gemm_wmma(
    const void* __restrict__ Av, int lda,
    const void* __restrict__ Bv, int ldb,
    void* __restrict__ Cv, int ldc, int K) {
    __shared__ __align__(16) unsigned short sA[BM * LDS_STRIDE];
    __shared__ __align__(16) unsigned short sB[BN * LDS_STRIDE];

    const int tid  = threadIdx.x;
    const int lane = tid & 31;
    const int wid  = tid >> 5;       // 8 waves: 4 (M) x 2 (N)
    const int wm   = wid >> 1;
    const int wn   = wid & 1;
    const int l15  = lane & 15;
    const int lh   = (lane >> 4) & 1;

    const int m0 = blockIdx.y * BM;
    const int n0 = blockIdx.x * BN;

    const int a_row  = tid >> 1;            // 2 threads per row, 16 elements each
    const int a_kseg = (tid & 1) * 16;
    const int b_k    = tid >> 3;            // B_F32 map: 8 threads per k-row
    const int b_nseg = (tid & 7) * 8;

    v8f acc[2][2] = {};

    for (int k0 = 0; k0 < K; k0 += BK) {
        // ---- stage A tile (BM x BK) ----
        if constexpr (AM == A_F32) {
            const float* A = (const float*)Av;
            cvt_store16(reinterpret_cast<const f32x4*>(
                            A + (size_t)(m0 + a_row) * lda + (k0 + a_kseg)),
                        &sA[a_row * LDS_STRIDE + a_kseg]);
            if (k0 + BK < K)
                __builtin_prefetch(A + (size_t)(m0 + a_row) * lda + (k0 + BK + a_kseg), 0, 1);
        } else {
            const unsigned short* A = (const unsigned short*)Av;
            stage32B(A + (size_t)(m0 + a_row) * lda + (k0 + a_kseg),
                     &sA[a_row * LDS_STRIDE + a_kseg]);
        }

        // ---- stage B tile into N-major LDS sB[n][k] ----
        if constexpr (BMODE == B_F32) {
            const float* Bp = (const float*)Bv;
            const float* pb = Bp + (size_t)(k0 + b_k) * ldb + (n0 + b_nseg);
#pragma unroll
            for (int i = 0; i < 8; ++i)
                sB[(b_nseg + i) * LDS_STRIDE + b_k] = f32_to_bf16(pb[i]);
        } else {
            if (tid < 128) {  // wave-uniform (waves 0..3)
                const unsigned short* Bt = (const unsigned short*)Bv;
                const int bn  = tid >> 1;
                const int bks = (tid & 1) * 16;
                stage32B(Bt + (size_t)(n0 + bn) * ldb + (k0 + bks),
                         &sB[bn * LDS_STRIDE + bks]);
            }
        }

#ifdef USE_ASYNC_LDS
        if constexpr (AM == A_BF16 || BMODE == B_BF16T)
            __builtin_amdgcn_s_wait_asynccnt(0);
#endif
        __syncthreads();

        // ---- fragments ----
        // A 16x32 bf16: lanes 0-15 row=l15, K {0-7,16-23}; lanes 16-31 K {8-15,24-31}
        Frag fa[2];
#pragma unroll
        for (int i = 0; i < 2; ++i) {
            const unsigned short* ab = &sA[(wm * 32 + i * 16 + l15) * LDS_STRIDE];
            fa[i].q[0] = *reinterpret_cast<const u32x4*>(ab + lh * 8);
            fa[i].q[1] = *reinterpret_cast<const u32x4*>(ab + 16 + lh * 8);
        }
        // B 32x16 bf16: lanes 0-15 col=l15, K {0-7,16-23}; lanes 16-31 K {8-15,24-31}
        Frag fb[2];
#pragma unroll
        for (int j = 0; j < 2; ++j) {
            const unsigned short* bb = &sB[(wn * 32 + j * 16 + l15) * LDS_STRIDE];
            fb[j].q[0] = *reinterpret_cast<const u32x4*>(bb + lh * 8);
            fb[j].q[1] = *reinterpret_cast<const u32x4*>(bb + 16 + lh * 8);
        }

#pragma unroll
        for (int i = 0; i < 2; ++i)
#pragma unroll
            for (int j = 0; j < 2; ++j)
                acc[i][j] = __builtin_amdgcn_wmma_f32_16x16x32_bf16(
                    false, fa[i].v, false, fb[j].v, (short)0, acc[i][j], false, false);
        __syncthreads();
    }

    // ---- epilogue: lane holds col = l15(+16*wn tile), rows m..m+7 ----
#pragma unroll
    for (int i = 0; i < 2; ++i) {
#pragma unroll
        for (int j = 0; j < 2; ++j) {
            const int col = n0 + wn * 32 + j * 16 + l15;
            const int row = m0 + wm * 32 + i * 16 + lh * 8;
            if constexpr (OM == OUT_F32) {
                float* cp = (float*)Cv + (size_t)row * ldc + col;
#pragma unroll
                for (int r = 0; r < 8; ++r) cp[(size_t)r * ldc] = acc[i][j][r];
            } else if constexpr (OM == OUT_BF16_ROW_RELU) {
                unsigned short* C = (unsigned short*)Cv;
#pragma unroll
                for (int r = 0; r < 8; ++r)
                    C[(size_t)(row + r) * ldc + col] = f32_to_bf16(fmaxf(acc[i][j][r], 0.0f));
            } else {  // OUT_BF16_TRANS: CT[col][row..row+7], one 16B store
                unsigned short* CT = (unsigned short*)Cv;
                u32x4 w;
                w.x = pack2(acc[i][j][0], acc[i][j][1]);
                w.y = pack2(acc[i][j][2], acc[i][j][3]);
                w.z = pack2(acc[i][j][4], acc[i][j][5]);
                w.w = pack2(acc[i][j][6], acc[i][j][7]);
                *reinterpret_cast<u32x4*>(CT + (size_t)col * ldc + row) = w;
            }
        }
    }
}

// zcat [8192x128] -> z_mean f32 | z_log_std f32 (d_out tail) + z bf16 [8192x64]
__global__ __launch_bounds__(256) void split_z(const float* __restrict__ zcat,
                                               float* __restrict__ zmean,
                                               float* __restrict__ zlog,
                                               unsigned short* __restrict__ zb) {
    const int idx = blockIdx.x * 256 + threadIdx.x;
    const int r = idx >> 7;
    const int c = idx & 127;
    const float v = zcat[idx];
    if (c < 64) {
        zmean[r * 64 + c] = v;
        zb[r * 64 + c] = f32_to_bf16(v);
    } else {
        zlog[r * 64 + (c - 64)] = v;
    }
}

extern "C" void kernel_launch(void* const* d_in, const int* in_sizes, int n_in,
                              void* d_out, int out_size, void* d_ws, size_t ws_size,
                              hipStream_t stream) {
    (void)in_sizes; (void)n_in; (void)out_size; (void)ws_size;

    const float* x   = (const float*)d_in[0];   // [8192,512]
    const float* adj = (const float*)d_in[1];   // [8192,8192]
    const float* Wh  = (const float*)d_in[2];   // [512,256]
    const float* Wm  = (const float*)d_in[3];   // [256,64]
    const float* Wl  = (const float*)d_in[4];   // [256,64]
    float* out = (float*)d_out;
    char*  wsb = (char*)d_ws;

    const int N = 8192, DIN = 512, DH = 256, DZ = 64;

    unsigned short* supportT = (unsigned short*)(wsb);                    // [256,8192] bf16, 4 MiB
    unsigned short* hidden   = (unsigned short*)(wsb + (4u << 20));       // [8192,256] bf16, 4 MiB
    unsigned short* zsupT    = (unsigned short*)(wsb + (8u << 20));       // [128,8192] bf16, 2 MiB
    float*          zcat     = (float*)        (wsb + (10u << 20));       // [8192,128] f32,  4 MiB
    unsigned short* zb       = (unsigned short*)(wsb + (14u << 20));      // [8192,64]  bf16, 1 MiB

    float* recon   = out;
    float* zmean_o = out + (size_t)N * N;
    float* zlog_o  = zmean_o + (size_t)N * DZ;

    const dim3 blk(256);

    // 1) supportT = (x @ Wh)^T
    gemm_wmma<A_F32, B_F32, OUT_BF16_TRANS><<<dim3(DH / BN, N / BM), blk, 0, stream>>>(
        x, DIN, Wh, DH, supportT, N, DIN);
    // 2) hidden = relu(adj @ support)   (B from supportT, async-staged)
    gemm_wmma<A_F32, B_BF16T, OUT_BF16_ROW_RELU><<<dim3(DH / BN, N / BM), blk, 0, stream>>>(
        adj, N, supportT, N, hidden, DH, N);
    // 3) zsupT rows 0:64 = (hidden @ Wm)^T ; rows 64:128 = (hidden @ Wl)^T
    gemm_wmma<A_BF16, B_F32, OUT_BF16_TRANS><<<dim3(1, N / BM), blk, 0, stream>>>(
        hidden, DH, Wm, DZ, zsupT, N, DH);
    gemm_wmma<A_BF16, B_F32, OUT_BF16_TRANS><<<dim3(1, N / BM), blk, 0, stream>>>(
        hidden, DH, Wl, DZ, zsupT + (size_t)DZ * N, N, DH);
    // 4) zcat = adj @ zsup  (fused mean|logstd: single adj pass)
    gemm_wmma<A_F32, B_BF16T, OUT_F32><<<dim3(128 / BN, N / BM), blk, 0, stream>>>(
        adj, N, zsupT, N, zcat, 128, N);
    // 5) split zcat -> d_out tail + z bf16
    split_z<<<(N * 128) / 256, blk, 0, stream>>>(zcat, zmean_o, zlog_o, zb);
    // 6) recon = z @ z^T  (both operands async-staged bf16)
    gemm_wmma<A_BF16, B_BF16T, OUT_F32><<<dim3(N / BN, N / BM), blk, 0, stream>>>(
        zb, DZ, zb, DZ, recon, N, DZ);
}
